// GraphCON_GRAFF_86388972191754
// MI455X (gfx1250) — matrix operand order repaired and checked by reading, not compiled, not run on hardware
//
#include <hip/hip_runtime.h>
#include <hip/hip_bf16.h>

typedef __attribute__((ext_vector_type(16))) _Float16 v16h;
typedef __attribute__((ext_vector_type(8)))  _Float16 v8h;
typedef __attribute__((ext_vector_type(4)))  _Float16 v4h;
typedef __attribute__((ext_vector_type(8)))  float    v8f;

#define NFEAT   256
#define NHID    128
#define NCLASS  40
#define NCPAD   64      // dec weight rows padded to 4 n-tiles
#define NLAYERS 4
#define STEP_SIZE 0.25f
#define A_STRIDE 40     // padded f16 row stride for 16x32 LDS A-tile (80B rows: conflict-free, 16B aligned)

// ---------------- WMMA fragment helpers (CDNA5 v_wmma_f32_16x16x32_f16 layouts) ----------------

// A-matrix 16x32 f16 (ISA 7.12.2): lane L: m = L%16, half = L/16
//   element e = A[m][half*16 + e] (contiguous k, matching B)
__device__ __forceinline__ v16h read_a_frag(const _Float16* As, int lane) {
  const int m = lane & 15, half = lane >> 4;
  const _Float16* p = As + m * A_STRIDE + half * 16;
  v8h lo = *(const v8h*)(p);        // ds_load_b128
  v8h hi = *(const v8h*)(p + 8);    // ds_load_b128
  v16h a;
#pragma unroll
  for (int i = 0; i < 8; ++i) { a[i] = lo[i]; a[i + 8] = hi[i]; }
  return a;
}

// B-matrix 32x16 f16: lane L: n = L%16, half = L/16, element e = B[k0 + half*16 + e][n].
// All B matrices here are pre-converted to f16 with B[k][n] contiguous per lane:
// 16 consecutive f16 = two global_load_b128.
__device__ __forceinline__ v16h load_b_frag16(const _Float16* p16) {
  v8h lo = *(const v8h*)(p16);
  v8h hi = *(const v8h*)(p16 + 8);
  v16h b;
#pragma unroll
  for (int i = 0; i < 8; ++i) { b[i] = lo[i]; b[i + 8] = hi[i]; }
  return b;
}

// Stage one 16x32 f32 tile -> f16 LDS tile. Exactly 128 threads participate:
// thread t: row m = t/8, quarter q = t%8 -> one coalesced float4 load + one packed v4h DS store.
__device__ __forceinline__ void stage_a_tile(const float* __restrict__ src, int lda,
                                             _Float16* As, int t) {
  const int m = t >> 3, q = t & 7;
  float4 f = ((const float4*)(src + (size_t)m * lda))[q];
  v4h h;
  h[0] = (_Float16)f.x; h[1] = (_Float16)f.y;
  h[2] = (_Float16)f.z; h[3] = (_Float16)f.w;
  *(v4h*)(As + m * A_STRIDE + q * 4) = h;   // ds_store_b64
}

// ---------------- prep kernels ----------------

__global__ void k_init_deg(float* deg, int n) {
  int i = blockIdx.x * blockDim.x + threadIdx.x;
  if (i < n) deg[i] = 1.0f;                 // self loop
}

__global__ void k_deg_add(const int* col, float* deg, int e) {
  int i = blockIdx.x * blockDim.x + threadIdx.x;
  if (i < e) atomicAdd(&deg[col[i]], 1.0f);
}

__global__ void k_deg_fin(float* deg_dinv, float* selfw, int n) {
  int i = blockIdx.x * blockDim.x + threadIdx.x;
  if (i < n) {
    float d = deg_dinv[i];
    deg_dinv[i] = rsqrtf(d);                // in place: deg -> d^-1/2
    selfw[i]    = 1.0f / d;
  }
}

__global__ void k_edge_w(const int* row, const int* col, const float* dinv, float* ew, int e) {
  int i = blockIdx.x * blockDim.x + threadIdx.x;
  if (i < e) ew[i] = dinv[row[i]] * dinv[col[i]];
}

// w_star = 0.5 (W + W^T), converted to f16 once (symmetric -> B rows == B cols)
__global__ void k_wstar16(const float* W, _Float16* wst16) {
  int i = blockIdx.x * blockDim.x + threadIdx.x;
  if (i < NHID * NHID) {
    int r = i >> 7, c = i & (NHID - 1);
    wst16[i] = (_Float16)(0.5f * (W[r * NHID + c] + W[c * NHID + r]));
  }
}

__global__ void k_cvt_encW(const float* encW, _Float16* encW16) {
  int i = blockIdx.x * blockDim.x + threadIdx.x;
  if (i < NHID * NFEAT) encW16[i] = (_Float16)encW[i];
}

// decW padded to 64 rows (rows >= NCLASS are zero) so the dec GEMM has no branches
__global__ void k_cvt_decW(const float* decW, _Float16* decW16) {
  int i = blockIdx.x * blockDim.x + threadIdx.x;
  if (i < NCPAD * NHID) {
    int r = i >> 7;
    decW16[i] = (r < NCLASS) ? (_Float16)decW[i] : (_Float16)0.0f;
  }
}

// S = selfw[m] * X[m][:]
__global__ void k_init_S(const float* X, const float* selfw, float* S, int n32) {
  int g = blockIdx.x * blockDim.x + threadIdx.x;
  if (g < n32) {
    int m = g >> 5, q = g & 31;
    float w = selfw[m];
    float4 v = ((const float4*)(X + (size_t)m * NHID))[q];
    float4 o; o.x = w * v.x; o.y = w * v.y; o.z = w * v.z; o.w = w * v.w;
    ((float4*)(S + (size_t)m * NHID))[q] = o;
  }
}

// one wave per edge: S[row] += ew * X[col]; 32 lanes x float4 = 128 features
__global__ __launch_bounds__(256) void k_spmm(const int* row, const int* col, const float* ew,
                                              const float* X, float* S, int e) {
  int g = blockIdx.x * blockDim.x + threadIdx.x;
  int edge = g >> 5, lane = g & 31;
  if (edge >= e) return;
  int r = row[edge], c = col[edge];
  float w = ew[edge];
  float4 v = ((const float4*)(X + (size_t)c * NHID))[lane];
  float* out = S + (size_t)r * NHID + lane * 4;
  atomicAdd(out + 0, w * v.x);
  atomicAdd(out + 1, w * v.y);
  atomicAdd(out + 2, w * v.z);
  atomicAdd(out + 3, w * v.w);
}

// ---------------- WMMA GEMM kernels ----------------
// 256 threads = 8 waves; block owns a 16-row x 128-col strip, one 16x16 n-tile per wave.
// A-tile staged in LDS once per k-step (f32 -> packed f16), shared by all 8 waves.

__global__ __launch_bounds__(256) void k_enc_gemm(const float* __restrict__ x,
                                                  const _Float16* __restrict__ encW16,
                                                  const float* __restrict__ encb,
                                                  float* __restrict__ X) {
  __shared__ alignas(16) _Float16 As[16 * A_STRIDE];
  const int m0 = blockIdx.x * 16;
  const int lane = threadIdx.x & 31;
  const int wave = threadIdx.x >> 5;
  const int half = lane >> 4;
  const int n = wave * 16 + (lane & 15);
  v8f c = {};
#pragma unroll
  for (int k0 = 0; k0 < NFEAT; k0 += 32) {
    __syncthreads();
    if (threadIdx.x < 128)
      stage_a_tile(x + (size_t)m0 * NFEAT + k0, NFEAT, As, threadIdx.x);
    __syncthreads();
    v16h a = read_a_frag(As, lane);
    v16h b = load_b_frag16(encW16 + (size_t)n * NFEAT + k0 + half * 16);
    c = __builtin_amdgcn_wmma_f32_16x16x32_f16(false, a, false, b, (short)0, c, false, false);
  }
  float bias = encb[n];
#pragma unroll
  for (int r = 0; r < 8; ++r) {
    int m = m0 + half * 8 + r;
    float v = c[r] + bias;
    X[(size_t)m * NHID + n] = v > 0.f ? v : 0.f;   // relu
  }
}

// X = relu(2*X + STEP * S @ w_star), in place (each lane touches only its own X elements)
__global__ __launch_bounds__(256) void k_layer_gemm(const float* __restrict__ S,
                                                    const _Float16* __restrict__ wst16,
                                                    float* __restrict__ X) {
  __shared__ alignas(16) _Float16 As[16 * A_STRIDE];
  const int m0 = blockIdx.x * 16;
  const int lane = threadIdx.x & 31;
  const int wave = threadIdx.x >> 5;
  const int half = lane >> 4;
  const int n = wave * 16 + (lane & 15);
  v8f c = {};
#pragma unroll
  for (int k0 = 0; k0 < NHID; k0 += 32) {
    __syncthreads();
    if (threadIdx.x < 128)
      stage_a_tile(S + (size_t)m0 * NHID + k0, NHID, As, threadIdx.x);
    __syncthreads();
    v16h a = read_a_frag(As, lane);
    // w_star symmetric: B[k][n] = wst[n][k] -> contiguous row read
    v16h b = load_b_frag16(wst16 + (size_t)n * NHID + k0 + half * 16);
    c = __builtin_amdgcn_wmma_f32_16x16x32_f16(false, a, false, b, (short)0, c, false, false);
  }
#pragma unroll
  for (int r = 0; r < 8; ++r) {
    int m = m0 + half * 8 + r;
    size_t idx = (size_t)m * NHID + n;
    float v = 2.0f * X[idx] + STEP_SIZE * c[r];
    X[idx] = v > 0.f ? v : 0.f;
  }
}

// out = X @ dec_W^T + dec_b  (N padded to 64 -> 4 waves; waves with n >= 40 discard results)
__global__ __launch_bounds__(128) void k_dec_gemm(const float* __restrict__ X,
                                                  const _Float16* __restrict__ decW16,
                                                  const float* __restrict__ decb,
                                                  float* __restrict__ out) {
  __shared__ alignas(16) _Float16 As[16 * A_STRIDE];
  const int m0 = blockIdx.x * 16;
  const int lane = threadIdx.x & 31;
  const int wave = threadIdx.x >> 5;
  const int half = lane >> 4;
  const int n = wave * 16 + (lane & 15);   // 0..63
  v8f c = {};
#pragma unroll
  for (int k0 = 0; k0 < NHID; k0 += 32) {
    __syncthreads();
    stage_a_tile(X + (size_t)m0 * NHID + k0, NHID, As, threadIdx.x);  // all 128 threads
    __syncthreads();
    v16h a = read_a_frag(As, lane);
    v16h b = load_b_frag16(decW16 + (size_t)n * NHID + k0 + half * 16);  // zero-padded rows
    c = __builtin_amdgcn_wmma_f32_16x16x32_f16(false, a, false, b, (short)0, c, false, false);
  }
  if (n < NCLASS) {
    float bias = decb[n];
#pragma unroll
    for (int r = 0; r < 8; ++r) {
      int m = m0 + half * 8 + r;
      out[(size_t)m * NCLASS + n] = c[r] + bias;
    }
  }
}

// ---------------- host-side orchestration ----------------

static inline size_t align8(size_t v) { return (v + 7) & ~(size_t)7; }  // keep 32B alignment in floats

extern "C" void kernel_launch(void* const* d_in, const int* in_sizes, int n_in,
                              void* d_out, int out_size, void* d_ws, size_t ws_size,
                              hipStream_t stream) {
  const float* x    = (const float*)d_in[0];
  const int*   eidx = (const int*)d_in[1];
  const float* encW = (const float*)d_in[2];
  const float* encb = (const float*)d_in[3];
  const float* W    = (const float*)d_in[4];
  const float* decW = (const float*)d_in[5];
  const float* decb = (const float*)d_in[6];
  float* out = (float*)d_out;

  const int N = in_sizes[0] / NFEAT;      // 100000 (divisible by 16)
  const int E = in_sizes[1] / 2;          // 1600000
  const int* erow = eidx;
  const int* ecol = eidx + E;

  // workspace layout; offsets kept 32B aligned
  float* wsf = (float*)d_ws;
  size_t off = 0;
  float* X     = wsf + off; off = align8(off + (size_t)N * NHID);
  float* S     = wsf + off; off = align8(off + (size_t)N * NHID);
  float* dinv  = wsf + off; off = align8(off + (size_t)N);     // deg, then d^-1/2 in place
  float* selfw = wsf + off; off = align8(off + (size_t)N);
  float* ew    = wsf + off; off = align8(off + (size_t)E);
  _Float16* f16base = (_Float16*)(wsf + off);
  size_t hoff = 0;
  _Float16* encW16 = f16base + hoff; hoff += (size_t)NHID * NFEAT;
  _Float16* wst16  = f16base + hoff; hoff += (size_t)NHID * NHID;
  _Float16* decW16 = f16base + hoff; hoff += (size_t)NCPAD * NHID;
  (void)hoff; (void)ws_size; (void)n_in; (void)out_size;

  const int T = 256;
  // graph normalization + weight preparation
  k_init_deg<<<(N + T - 1) / T, T, 0, stream>>>(dinv, N);
  k_deg_add <<<(E + T - 1) / T, T, 0, stream>>>(ecol, dinv, E);
  k_deg_fin <<<(N + T - 1) / T, T, 0, stream>>>(dinv, selfw, N);
  k_edge_w  <<<(E + T - 1) / T, T, 0, stream>>>(erow, ecol, dinv, ew, E);
  k_wstar16 <<<(NHID * NHID + T - 1) / T, T, 0, stream>>>(W, wst16);
  k_cvt_encW<<<(NHID * NFEAT + T - 1) / T, T, 0, stream>>>(encW, encW16);
  k_cvt_decW<<<(NCPAD * NHID + T - 1) / T, T, 0, stream>>>(decW, decW16);

  const int mtiles = N / 16;              // 6250

  // encoder: X = relu(x @ encW^T + encb)
  k_enc_gemm<<<mtiles, 256, 0, stream>>>(x, encW16, encb, X);

  // 4 GraphCON-GRAFF layers (Y state cancels algebraically): X = relu(2X + STEP*spmm(X)@w_star)
  const int n32 = N * (NHID / 4);
  for (int l = 0; l < NLAYERS; ++l) {
    k_init_S<<<(n32 + T - 1) / T, T, 0, stream>>>(X, selfw, S, n32);
    k_spmm  <<<(E * 32 + T - 1) / T, T, 0, stream>>>(erow, ecol, ew, X, S, E);
    k_layer_gemm<<<mtiles, 256, 0, stream>>>(S, wst16, X);
  }

  // decoder
  k_dec_gemm<<<mtiles, 128, 0, stream>>>(X, decW16, decb, out);
}
